// SatTerrestrialCritic_60120952209437
// MI455X (gfx1250) — matrix operand order repaired and checked by reading, not compiled, not run on hardware
//
#include <hip/hip_runtime.h>
#include <hip/hip_bf16.h>
#include <stdint.h>

// CDNA5 wave32 WMMA types
typedef __attribute__((ext_vector_type(16))) __bf16 v16bf;
typedef __attribute__((ext_vector_type(8)))  float  v8f;

#define NEG_SLOPE 0.2f

__device__ __forceinline__ unsigned f2o(float f) {
  unsigned u = __float_as_uint(f);
  return (u & 0x80000000u) ? ~u : (u | 0x80000000u);
}
__device__ __forceinline__ float o2f(unsigned u) {
  u = (u & 0x80000000u) ? (u & 0x7FFFFFFFu) : ~u;
  return __uint_as_float(u);
}

// ---------------------------------------------------------------------------
// Repack fp32 weight matrix [K, NOUT] (row-major) into per-lane bf16 WMMA
// B-fragments: frag[((kk*NTtot + nt)*32 + lane)*16 + j]
//   lane = g*16 + n  (g = lane/16), element j <-> K = kk*32 + g*16 + j,
//   col = nt*16 + (lane%16)      (contiguous k order, same as the A fragments)
// ---------------------------------------------------------------------------
__global__ void gat_repack_kernel(const float* __restrict__ W, __bf16* __restrict__ frag,
                                  int KC, int NTtot, int NOUT) {
  int f = blockIdx.x * blockDim.x + threadIdx.x;
  int total = KC * NTtot * 32 * 16;
  if (f >= total) return;
  int j    = f & 15;
  int lane = (f >> 4) & 31;
  int rest = f >> 9;
  int nt = rest % NTtot;
  int kk = rest / NTtot;
  int k   = kk * 32 + (lane >> 4) * 16 + j;
  int col = nt * 16 + (lane & 15);
  frag[f] = (__bf16)W[(size_t)k * NOUT + col];
}

// ---------------------------------------------------------------------------
// Generic WMMA GEMM: out[N, NOUT] = A[N, K] @ W  (bf16 inputs, fp32 accum)
// One wave handles a 16-row tile x NT 16-col tiles. Requires N % 16 == 0
// (N = 50000 = 3125 * 16 here).
// A-fragment per-lane layout (contiguous k order, same as the B repack): lane = half*16 + m holds
// row m; element j <-> local K = half*16 + j.
// D layout: lane holds rows half*8 + r (r=0..7), col = lane%16.
// ---------------------------------------------------------------------------
template <int KC, int NT, int NOUT, bool CONCAT, bool BIAS>
__global__ void gat_gemm_wmma_kernel(const float* __restrict__ A0,
                                     const float* __restrict__ A1,
                                     const __bf16* __restrict__ frag,
                                     const float* __restrict__ bias,
                                     float* __restrict__ out,
                                     int nrows) {
  const int lane = threadIdx.x & 31;
  const int wave = threadIdx.x >> 5;
  const int tile = blockIdx.x * (blockDim.x >> 5) + wave;
  const int row0 = tile * 16;
  if (row0 >= nrows) return;
  const int half = lane >> 4;
  const int mrow = lane & 15;
  const int row  = row0 + mrow;
  const int colBase = blockIdx.y * NT;   // in 16-col tiles
  const int NTtot = NOUT / 16;

  v8f acc[NT];
#pragma unroll
  for (int nt = 0; nt < NT; ++nt) { v8f z = {}; acc[nt] = z; }

#pragma unroll
  for (int kk = 0; kk < KC; ++kk) {
    v16bf a;
#pragma unroll
    for (int j = 0; j < 16; ++j) {
      int klocal = half * 16 + j;
      int k = kk * 32 + klocal;
      float v;
      if (CONCAT) {  // A = [node_features(96) | actions(32)]
        v = (k < 96) ? A0[(size_t)row * 96 + k] : A1[(size_t)row * 32 + (k - 96)];
      } else {
        v = A0[(size_t)row * (KC * 32) + k];
      }
      a[j] = (__bf16)v;
    }
#pragma unroll
    for (int nt = 0; nt < NT; ++nt) {
      int ct = colBase + nt;
      const v16bf* bp =
          (const v16bf*)(frag + ((size_t)(kk * NTtot + ct) * 32 + lane) * 16);
      v16bf b = *bp;
      acc[nt] = __builtin_amdgcn_wmma_f32_16x16x32_bf16(
          false, a, false, b, (short)0, acc[nt], false, false);
    }
  }

#pragma unroll
  for (int nt = 0; nt < NT; ++nt) {
    int col = (colBase + nt) * 16 + mrow;
    float bv = BIAS ? bias[col] : 0.0f;
#pragma unroll
    for (int r = 0; r < 8; ++r) {
      int orow = row0 + half * 8 + r;
      out[(size_t)orow * NOUT + col] = acc[nt][r] + bv;
    }
  }
}

// ---------------------------------------------------------------------------
// Per-(node, head) attention logits: a_s = <h[n,h,:], att_s[h,:]>, same a_d.
// ---------------------------------------------------------------------------
__global__ void gat_attn_logits_kernel(const float* __restrict__ h,
                                       const float* __restrict__ att_s,
                                       const float* __restrict__ att_d,
                                       float* __restrict__ a_s, float* __restrict__ a_d,
                                       int n, int heads, int hid) {
  int i = blockIdx.x * blockDim.x + threadIdx.x;
  if (i >= n * heads) return;
  int node = i / heads, hd = i % heads;
  const float* hp = h + (size_t)node * heads * hid + (size_t)hd * hid;
  const float* as = att_s + hd * hid;
  const float* ad = att_d + hd * hid;
  float ss = 0.f, dd = 0.f;
  for (int c = 0; c < hid; ++c) { float v = hp[c]; ss += v * as[c]; dd += v * ad[c]; }
  a_s[i] = ss;
  a_d[i] = dd;
}

// ---------------------------------------------------------------------------
// Edge pass A: per-dst segment max of leaky_relu(a_s[src]+a_d[dst]) via
// order-preserving uint atomicMax (m memset to 0 == below every finite code).
// Edges E..E+n-1 are the implicit self-loops.
// ---------------------------------------------------------------------------
__global__ void gat_edge_max_kernel(const int* __restrict__ ei, int E, int n, int heads,
                                    const float* __restrict__ a_s,
                                    const float* __restrict__ a_d,
                                    unsigned* __restrict__ m) {
  long long i = (long long)blockIdx.x * blockDim.x + threadIdx.x;
  long long total = (long long)(E + n) * heads;
  if (i >= total) return;
  int h = (int)(i % heads);
  long long e = i / heads;
  int s, d;
  if (e < E) { s = ei[e]; d = ei[E + e]; } else { s = d = (int)(e - E); }
  float v = a_s[(size_t)s * heads + h] + a_d[(size_t)d * heads + h];
  v = (v > 0.f) ? v : NEG_SLOPE * v;
  atomicMax(&m[(size_t)d * heads + h], f2o(v));
}

// ---------------------------------------------------------------------------
// Edge pass B (fused): ex = exp(lrelu - m[dst]); denom[dst] += ex;
// acc[dst, :] += ex * h[src, :].  One wave per edge, C/32 channels per lane.
// ---------------------------------------------------------------------------
template <int HEADS, int HID>
__global__ void gat_edge_agg_kernel(const int* __restrict__ ei, int E, int n,
                                    const float* __restrict__ a_s,
                                    const float* __restrict__ a_d,
                                    const unsigned* __restrict__ m,
                                    const float* __restrict__ h,
                                    float* __restrict__ denom,
                                    float* __restrict__ acc) {
  const int C = HEADS * HID;
  const int CPL = C / 32;
  int lane = threadIdx.x & 31;
  long long wid = (long long)blockIdx.x * (blockDim.x >> 5) + (threadIdx.x >> 5);
  long long tot = (long long)E + n;
  if (wid >= tot) return;
  int s, d;
  if (wid < E) { s = ei[wid]; d = ei[E + wid]; } else { s = d = (int)(wid - E); }
  int c0 = lane * CPL;
  int hd = c0 / HID;
  float v = a_s[(size_t)s * HEADS + hd] + a_d[(size_t)d * HEADS + hd];
  v = (v > 0.f) ? v : NEG_SLOPE * v;
  float ex = __expf(v - o2f(m[(size_t)d * HEADS + hd]));
  if ((c0 % HID) == 0) atomicAdd(&denom[(size_t)d * HEADS + hd], ex);
  const float* hs = h + (size_t)s * C;
  float* ap = acc + (size_t)d * C;
#pragma unroll
  for (int j = 0; j < CPL; ++j) atomicAdd(&ap[c0 + j], ex * hs[c0 + j]);
}

// out = relu(acc / (denom + eps) + bias), in place
__global__ void gat_node_finalize_kernel(float* __restrict__ acc,
                                         const float* __restrict__ den,
                                         const float* __restrict__ bias,
                                         int n, int heads, int hid) {
  long long i = (long long)blockIdx.x * blockDim.x + threadIdx.x;
  long long total = (long long)n * heads * hid;
  if (i >= total) return;
  int C = heads * hid;
  int c = (int)(i % C);
  long long node = i / C;
  int hd = c / hid;
  float v = acc[i] / (den[node * heads + hd] + 1e-16f) + bias[c];
  acc[i] = (v > 0.f) ? v : 0.f;
}

// Layer-2 finalize fused with column-sum for the global mean (64 channels).
__global__ void gat_finalize2_mean_kernel(const float* __restrict__ acc2,
                                          const float* __restrict__ den2,
                                          const float* __restrict__ b2,
                                          float* __restrict__ gsum, int n) {
  __shared__ float red[256];
  int t = threadIdx.x;
  int c = t & 63;
  int part = t >> 6;  // 0..3
  float s = 0.f;
  for (int node = blockIdx.x * 4 + part; node < n; node += gridDim.x * 4) {
    float v = acc2[(size_t)node * 64 + c] / (den2[node] + 1e-16f) + b2[c];
    s += (v > 0.f) ? v : 0.f;
  }
  red[t] = s;
  __syncthreads();
  if (t < 64) atomicAdd(&gsum[c], red[t] + red[64 + t] + red[128 + t] + red[192 + t]);
}

// out[0] = (gsum/N) . W_v + b_v
__global__ void gat_final_dot_kernel(const float* __restrict__ gsum,
                                     const float* __restrict__ Wv,
                                     const float* __restrict__ bv,
                                     float* __restrict__ out, int n) {
  __shared__ float red[64];
  int t = threadIdx.x;  // 64 threads
  red[t] = (gsum[t] / (float)n) * Wv[t];
  __syncthreads();
  if (t == 0) {
    float s = 0.f;
    for (int i = 0; i < 64; ++i) s += red[i];
    out[0] = s + bv[0];
  }
}

// ---------------------------------------------------------------------------
extern "C" void kernel_launch(void* const* d_in, const int* in_sizes, int n_in,
                              void* d_out, int out_size, void* d_ws, size_t ws_size,
                              hipStream_t stream) {
  (void)n_in; (void)out_size; (void)ws_size;
  const float* nf   = (const float*)d_in[0];   // [N,96]
  const float* act  = (const float*)d_in[1];   // [N,32]
  const int*   ei   = (const int*)d_in[2];     // [2,E] (src row, dst row)
  const float* Wemb = (const float*)d_in[3];   // [128,64]
  const float* bemb = (const float*)d_in[4];   // [64]
  const float* W1   = (const float*)d_in[5];   // [64,256]
  const float* as1w = (const float*)d_in[6];   // [4,64]
  const float* ad1w = (const float*)d_in[7];   // [4,64]
  const float* b1   = (const float*)d_in[8];   // [256]
  const float* W2   = (const float*)d_in[9];   // [256,64]
  const float* as2w = (const float*)d_in[10];  // [1,64]
  const float* ad2w = (const float*)d_in[11];  // [1,64]
  const float* b2   = (const float*)d_in[12];  // [64]
  const float* Wv   = (const float*)d_in[13];  // [64,1]
  const float* bv   = (const float*)d_in[14];  // [1]

  const int N = in_sizes[0] / 96;
  const int E = in_sizes[2] / 2;

  // ---- workspace layout (zero-init region first -> single memset node) ----
  char* ws = (char*)d_ws;
  size_t o = 0;
  auto alloc = [&](size_t bytes) -> size_t {
    size_t r = o; o = (o + bytes + 255) & ~(size_t)255; return r;
  };
  size_t off_acc1 = alloc((size_t)N * 256 * 4);  // layer1 numerator -> out1
  size_t off_acc2 = alloc((size_t)N * 64 * 4);   // layer2 numerator
  size_t off_den1 = alloc((size_t)N * 4 * 4);
  size_t off_den2 = alloc((size_t)N * 4);
  size_t off_m1   = alloc((size_t)N * 4 * 4);    // ordered-uint maxes
  size_t off_m2   = alloc((size_t)N * 4);
  size_t off_gsum = alloc(64 * 4);
  size_t zeroBytes = o;                          // everything above zeroed per call
  size_t off_xemb = alloc((size_t)N * 64 * 4);   // x_emb, later reused for h2
  size_t off_h1   = alloc((size_t)N * 256 * 4);
  size_t off_as1  = alloc((size_t)N * 4 * 4);
  size_t off_ad1  = alloc((size_t)N * 4 * 4);
  size_t off_as2  = alloc((size_t)N * 4);
  size_t off_ad2  = alloc((size_t)N * 4);
  size_t off_fWe  = alloc((size_t)4 * 4 * 32 * 16 * 2);   // KC=4, NT=4
  size_t off_fW1  = alloc((size_t)2 * 16 * 32 * 16 * 2);  // KC=2, NT=16
  size_t off_fW2  = alloc((size_t)8 * 4 * 32 * 16 * 2);   // KC=8, NT=4

  float*    acc1 = (float*)(ws + off_acc1);
  float*    acc2 = (float*)(ws + off_acc2);
  float*    den1 = (float*)(ws + off_den1);
  float*    den2 = (float*)(ws + off_den2);
  unsigned* m1   = (unsigned*)(ws + off_m1);
  unsigned* m2   = (unsigned*)(ws + off_m2);
  float*    gsum = (float*)(ws + off_gsum);
  float*    xemb = (float*)(ws + off_xemb);    // also h2
  float*    h1   = (float*)(ws + off_h1);
  float*    as1  = (float*)(ws + off_as1);
  float*    ad1  = (float*)(ws + off_ad1);
  float*    as2  = (float*)(ws + off_as2);
  float*    ad2  = (float*)(ws + off_ad2);
  __bf16*   fWe  = (__bf16*)(ws + off_fWe);
  __bf16*   fW1  = (__bf16*)(ws + off_fW1);
  __bf16*   fW2  = (__bf16*)(ws + off_fW2);

  hipMemsetAsync(d_ws, 0, zeroBytes, stream);

  // ---- weight repack to bf16 WMMA fragments (cheap, per call) ----
  gat_repack_kernel<<<(4 * 4 * 32 * 16 + 255) / 256, 256, 0, stream>>>(Wemb, fWe, 4, 4, 64);
  gat_repack_kernel<<<(2 * 16 * 32 * 16 + 255) / 256, 256, 0, stream>>>(W1, fW1, 2, 16, 256);
  gat_repack_kernel<<<(8 * 4 * 32 * 16 + 255) / 256, 256, 0, stream>>>(W2, fW2, 8, 4, 64);

  const int tiles = (N + 15) / 16;            // 3125 (N multiple of 16)
  const int gemmBlocks = (tiles + 7) / 8;     // 8 waves/block

  // ---- x_emb = [nf|act] @ W_emb + b_emb  (K=128, NOUT=64) ----
  gat_gemm_wmma_kernel<4, 4, 64, true, true>
      <<<dim3(gemmBlocks, 1), 256, 0, stream>>>(nf, act, fWe, bemb, xemb, N);

  // ---- h1 = x_emb @ W1  (K=64, NOUT=256; split cols across gridDim.y) ----
  gat_gemm_wmma_kernel<2, 8, 256, false, false>
      <<<dim3(gemmBlocks, 2), 256, 0, stream>>>(xemb, nullptr, fW1, nullptr, h1, N);

  // ---- layer 1: GAT edge softmax + aggregate ----
  gat_attn_logits_kernel<<<(N * 4 + 255) / 256, 256, 0, stream>>>(h1, as1w, ad1w, as1, ad1, N, 4, 64);
  {
    long long tot = (long long)(E + N) * 4;
    gat_edge_max_kernel<<<(unsigned)((tot + 255) / 256), 256, 0, stream>>>(ei, E, N, 4, as1, ad1, m1);
  }
  gat_edge_agg_kernel<4, 64><<<(E + N + 7) / 8, 256, 0, stream>>>(ei, E, N, as1, ad1, m1, h1, den1, acc1);
  {
    long long tot = (long long)N * 256;
    gat_node_finalize_kernel<<<(unsigned)((tot + 255) / 256), 256, 0, stream>>>(acc1, den1, b1, N, 4, 64);
  }

  // ---- h2 = out1 @ W2  (K=256, NOUT=64); reuse x_emb buffer for h2 ----
  gat_gemm_wmma_kernel<8, 4, 64, false, false>
      <<<dim3(gemmBlocks, 1), 256, 0, stream>>>(acc1, nullptr, fW2, nullptr, xemb, N);

  // ---- layer 2 (H=1) ----
  gat_attn_logits_kernel<<<(N + 255) / 256, 256, 0, stream>>>(xemb, as2w, ad2w, as2, ad2, N, 1, 64);
  gat_edge_max_kernel<<<(E + N + 255) / 256, 256, 0, stream>>>(ei, E, N, 1, as2, ad2, m2);
  gat_edge_agg_kernel<1, 64><<<(E + N + 7) / 8, 256, 0, stream>>>(ei, E, N, as2, ad2, m2, xemb, den2, acc2);

  // ---- finalize layer 2 fused with global mean, then head dot ----
  gat_finalize2_mean_kernel<<<256, 256, 0, stream>>>(acc2, den2, b2, gsum, N);
  gat_final_dot_kernel<<<1, 64, 0, stream>>>(gsum, Wv, bv, (float*)d_out, N);
}
